// Only_Audio_Guided_Attention_55370718380568
// MI455X (gfx1250) — matrix operand order repaired and checked by reading, not compiled, not run on hardware
//
#include <hip/hip_runtime.h>
#include <hip/hip_bf16.h>

typedef __attribute__((ext_vector_type(16))) __bf16 v16bf;
typedef __attribute__((ext_vector_type(8)))  float  v8f;

#define B_   256
#define T_   10
#define BT_  2560          // B*T
#define NSP  49            // H*W
#define TOK  125440        // BT*49
#define VD_  512
#define AD_  128
#define HD_  256
#define QD_  128

// ---------------------------------------------------------------------------
// Generic WMMA GEMM:  Y[M,N] = act(X[M,K] @ W[N,K]^T + bias[N]) [* aux]
// WG = 256 threads = 8 waves, tile 128(M) x 64(N); each wave owns a 32x32
// output block = 2x2 16x16 accumulators (A/B fragments reused twice each).
// K stepped by 32.
// ---------------------------------------------------------------------------
template<bool XBF16, bool OUTBF16, int ACT, bool SCALE49>
__global__ __launch_bounds__(256)
void gemm_wmma(const void* __restrict__ Xv, const float* __restrict__ W,
               const float* __restrict__ bias, void* __restrict__ Yv,
               const float* __restrict__ aux, int M, int N, int K) {
  __shared__ __bf16 Xt[128][48];  // pitch 48 keeps 16B alignment, skews banks
  __shared__ __bf16 Wt[64][48];
  const int tid  = threadIdx.x;
  const int lane = tid & 31;
  const int wave = tid >> 5;
  const int wy   = wave >> 1;          // 0..3  (M sub-tile, 32 rows each)
  const int wx   = wave & 1;           // 0..1  (N sub-tile, 32 cols each)
  const int mBase = blockIdx.y * 128;
  const int nBase = blockIdx.x * 64;
  const float*  Xf = (const float*)Xv;
  const __bf16* Xh = (const __bf16*)Xv;

  v8f acc[2][2] = {};
  for (int kk = 0; kk < K; kk += 32) {
    { // stage X tile 128x32 (16 elems/thread, coalesced along K)
      int r  = tid >> 1;
      int c0 = (tid & 1) * 16;
      long g = (long)(mBase + r) * K + kk + c0;
      if (XBF16) {
        #pragma unroll
        for (int i = 0; i < 16; ++i) Xt[r][c0 + i] = Xh[g + i];
      } else {
        #pragma unroll
        for (int i = 0; i < 16; ++i) Xt[r][c0 + i] = (__bf16)Xf[g + i];
      }
    }
    { // stage W tile 64(N)x32(K) (8 elems/thread)
      int r  = tid >> 2;
      int c0 = (tid & 3) * 8;
      long g = (long)(nBase + r) * K + kk + c0;
      #pragma unroll
      for (int i = 0; i < 8; ++i) Wt[r][c0 + i] = (__bf16)W[g + i];
    }
    __syncthreads();
    // gather fragments in contiguous k order (k = 16*(lane/16) + s), same for A and B
    const int rl = lane & 15;
    const int kA = (lane >= 16) ? 16 : 0;
    const int kB = (lane >= 16) ? 16 : 0;
    v16bf af[2], bfr[2];
    #pragma unroll
    for (int i = 0; i < 2; ++i) {
      int row = wy * 32 + i * 16 + rl;
      #pragma unroll
      for (int j = 0; j < 8; ++j) af[i][j]     = Xt[row][kA + j];
      #pragma unroll
      for (int j = 0; j < 8; ++j) af[i][8 + j] = Xt[row][kA + 8 + j];
      int n = wx * 32 + i * 16 + rl;
      #pragma unroll
      for (int j = 0; j < 16; ++j) bfr[i][j] = Wt[n][kB + j];
    }
    #pragma unroll
    for (int i = 0; i < 2; ++i)
      #pragma unroll
      for (int j = 0; j < 2; ++j)
        acc[i][j] = __builtin_amdgcn_wmma_f32_16x16x32_bf16(
            false, af[i], false, bfr[j], (short)0, acc[i][j], false, false);
    __syncthreads();
  }
  // epilogue: C layout — element r: M = r + (lane>=16 ? 8 : 0), N = lane%16
  #pragma unroll
  for (int j = 0; j < 2; ++j) {
    int col = nBase + wx * 32 + j * 16 + (lane & 15);
    float bv = bias[col];
    #pragma unroll
    for (int i = 0; i < 2; ++i) {
      #pragma unroll
      for (int r = 0; r < 8; ++r) {
        int row = mBase + wy * 32 + i * 16 + r + ((lane >= 16) ? 8 : 0);
        float y = acc[i][j][r] + bv;
        if (ACT == 1) y = fmaxf(y, 0.0f);
        else if (ACT == 2) y = 1.0f / (1.0f + __expf(-y));
        if (SCALE49) y *= aux[(long)(row / NSP) * N + col];
        if (OUTBF16) ((__bf16*)Yv)[(long)row * N + col] = (__bf16)y;
        else         ((float*)Yv)[(long)row * N + col]  = y;
      }
    }
  }
}

// ---------------------------------------------------------------------------
// audio [T,B,AD] -> [B*T, AD]  (this IS output #2, also reused as GEMM input)
// ---------------------------------------------------------------------------
__global__ void audio_tr_kernel(const float* __restrict__ audio,
                                float* __restrict__ out) {
  long i = (long)blockIdx.x * 256 + threadIdx.x;     // BT*AD
  int bt = (int)(i >> 7), c = (int)(i & 127);
  int b = bt / T_, t = bt % T_;
  out[i] = audio[((long)t * B_ + b) * AD_ + c];
}

// ---------------------------------------------------------------------------
// Per-batch: S = softmax(q k^T), out = S v, vf2 = LN(vf + out) -> bf16,
// plus spatial mean of vf2 (/49).  One WG of 256 per batch, dynamic LDS.
// ---------------------------------------------------------------------------
__global__ __launch_bounds__(256)
void attn_ln_kernel(const float* __restrict__ vf, const float* __restrict__ Qb,
                    const float* __restrict__ Kb, const float* __restrict__ Vb,
                    const float* __restrict__ ln_g, const float* __restrict__ ln_b,
                    __bf16* __restrict__ vf2, float* __restrict__ vf2avg) {
  extern __shared__ float sm[];
  float* sq  = sm;                       // 49*128
  float* sk  = sq + NSP * QD_;           // 49*128
  float* sv  = sk + NSP * QD_;           // 49*512
  float* ss  = sv + NSP * VD_;           // 49*50
  float* red = ss + NSP * 50;            // 256
  const int b = blockIdx.x, tid = threadIdx.x;
  const long t0 = (long)b * NSP;

  for (int i = tid; i < NSP * QD_; i += 256) {
    int n = i >> 7, c = i & 127;
    sq[i] = Qb[(t0 + n) * QD_ + c];
    sk[i] = Kb[(t0 + n) * QD_ + c];
  }
  for (int i = tid; i < NSP * VD_; i += 256)
    sv[i] = Vb[t0 * VD_ + i];
  __syncthreads();

  for (int i = tid; i < NSP * NSP; i += 256) {
    int n = i / NSP, m = i % NSP;
    float s = 0.f;
    for (int c = 0; c < QD_; ++c) s += sq[n * QD_ + c] * sk[m * QD_ + c];
    ss[n * 50 + m] = s;
  }
  __syncthreads();
  if (tid < NSP) {
    float mx = -1e30f;
    for (int m = 0; m < NSP; ++m) mx = fmaxf(mx, ss[tid * 50 + m]);
    float sum = 0.f;
    for (int m = 0; m < NSP; ++m) {
      float e = __expf(ss[tid * 50 + m] - mx);
      ss[tid * 50 + m] = e; sum += e;
    }
    float inv = 1.0f / sum;
    for (int m = 0; m < NSP; ++m) ss[tid * 50 + m] *= inv;
  }
  __syncthreads();

  float cs0 = 0.f, cs1 = 0.f;
  for (int n = 0; n < NSP; ++n) {
    float a0 = 0.f, a1 = 0.f;
    for (int m = 0; m < NSP; ++m) {
      float w = ss[n * 50 + m];
      a0 += w * sv[m * VD_ + tid];
      a1 += w * sv[m * VD_ + tid + 256];
    }
    float y0 = a0 + vf[(t0 + n) * VD_ + tid];
    float y1 = a1 + vf[(t0 + n) * VD_ + tid + 256];
    red[tid] = y0 + y1;
    __syncthreads();
    for (int s = 128; s > 0; s >>= 1) { if (tid < s) red[tid] += red[tid + s]; __syncthreads(); }
    float mu = red[0] * (1.0f / VD_);
    __syncthreads();
    float d0 = y0 - mu, d1 = y1 - mu;
    red[tid] = d0 * d0 + d1 * d1;
    __syncthreads();
    for (int s = 128; s > 0; s >>= 1) { if (tid < s) red[tid] += red[tid + s]; __syncthreads(); }
    float rstd = rsqrtf(red[0] * (1.0f / VD_) + 1e-5f);
    __syncthreads();
    float o0 = d0 * rstd * ln_g[tid]       + ln_b[tid];
    float o1 = d1 * rstd * ln_g[tid + 256] + ln_b[tid + 256];
    vf2[(t0 + n) * VD_ + tid]       = (__bf16)o0;
    vf2[(t0 + n) * VD_ + tid + 256] = (__bf16)o1;
    cs0 += o0; cs1 += o1;
  }
  vf2avg[(long)b * VD_ + tid]       = cs0 * (1.0f / NSP);
  vf2avg[(long)b * VD_ + tid + 256] = cs1 * (1.0f / NSP);
}

// mean over the 49 spatial positions of prod -> [BT, VD]
__global__ void mean49_kernel(const __bf16* __restrict__ prod,
                              float* __restrict__ out) {
  long i = (long)blockIdx.x * 256 + threadIdx.x;   // BT*VD
  int b = (int)(i >> 9), c = (int)(i & 511);
  long base = ((long)b * NSP) * VD_ + c;
  float s = 0.f;
  for (int n = 0; n < NSP; ++n) s += (float)prod[base + (long)n * VD_];
  out[i] = s * (1.0f / NSP);
}

// c_att (bf16) = video * (ch + 1)
__global__ void catt_kernel(const float* __restrict__ video,
                            const float* __restrict__ ch,
                            __bf16* __restrict__ catt) {
  long i = (long)blockIdx.x * 256 + threadIdx.x;   // TOK*VD
  long t = i >> 9; int c = (int)(i & 511);
  int b = (int)(t / NSP);
  catt[i] = (__bf16)(video[i] * (ch[(long)b * VD_ + c] + 1.0f));
}

// fused rank-1 projections: tanh(b + sum_h x[t,h]*g[b,h]*w[h]) for both branches
__global__ void score_kernel(const __bf16* __restrict__ relu3,
                             const float* __restrict__ vavg,
                             const float* __restrict__ w_avatt, const float* __restrict__ b_avatt,
                             const __bf16* __restrict__ cq,
                             const float* __restrict__ aq2,
                             const float* __restrict__ w_sa, const float* __restrict__ b_sa,
                             float* __restrict__ s_self, float* __restrict__ s_sp) {
  long t = (long)blockIdx.x * 256 + threadIdx.x;   // TOK
  int b = (int)(t / NSP);
  long r = t * HD_;
  long vb = (long)b * HD_;
  float s1 = b_avatt[0], s2 = b_sa[0];
  for (int h = 0; h < HD_; ++h) {
    s1 += (float)relu3[r + h] * vavg[vb + h] * w_avatt[h];
    s2 += (float)cq[r + h]   * aq2[vb + h]  * w_sa[h];
  }
  s_self[t] = tanhf(s1);
  s_sp[t]   = tanhf(s2);
}

// per-batch: dual softmax over 49, weighted sums, (1 + 0.5*sigmoid) scaling
__global__ __launch_bounds__(256)
void final_kernel(const float* __restrict__ video, const float* __restrict__ ch,
                  const __bf16* __restrict__ vf2,
                  const float* __restrict__ s_self, const float* __restrict__ s_sp,
                  float* __restrict__ out) {
  __shared__ float m1[NSP], m2[NSP];
  int b = blockIdx.x, tid = threadIdx.x;
  if (tid == 0) {
    float mx = -1e30f;
    for (int n = 0; n < NSP; ++n) mx = fmaxf(mx, s_self[b * NSP + n]);
    float sum = 0.f;
    for (int n = 0; n < NSP; ++n) { float e = __expf(s_self[b * NSP + n] - mx); m1[n] = e; sum += e; }
    for (int n = 0; n < NSP; ++n) m1[n] /= sum;
    mx = -1e30f;
    for (int n = 0; n < NSP; ++n) mx = fmaxf(mx, s_sp[b * NSP + n]);
    sum = 0.f;
    for (int n = 0; n < NSP; ++n) { float e = __expf(s_sp[b * NSP + n] - mx); m2[n] = e; sum += e; }
    for (int n = 0; n < NSP; ++n) m2[n] /= sum;
  }
  __syncthreads();
  for (int c = tid; c < VD_; c += 256) {
    float chv = ch[(long)b * VD_ + c] + 1.0f;
    long base = (long)b * NSP * VD_ + c;
    float sa = 0.f, cs = 0.f;
    for (int n = 0; n < NSP; ++n) {
      sa += m1[n] * (float)vf2[base + (long)n * VD_];
      cs += m2[n] * (video[base + (long)n * VD_] * chv);
    }
    cs *= 1.0f + 0.5f / (1.0f + __expf(-sa));
    out[(long)b * VD_ + c] = cs;
  }
}

// ---------------------------------------------------------------------------
extern "C" void kernel_launch(void* const* d_in, const int* in_sizes, int n_in,
                              void* d_out, int out_size, void* d_ws, size_t ws_size,
                              hipStream_t stream) {
  const float* video  = (const float*)d_in[0];
  const float* audio  = (const float*)d_in[1];
  const float* wq = (const float*)d_in[2],  *bq = (const float*)d_in[3];
  const float* wk = (const float*)d_in[4],  *bk = (const float*)d_in[5];
  const float* wv = (const float*)d_in[6],  *bv = (const float*)d_in[7];
  const float* ln_g = (const float*)d_in[8], *ln_b = (const float*)d_in[9];
  const float* w_ave=(const float*)d_in[10], *b_ave=(const float*)d_in[11];
  const float* w_v3 =(const float*)d_in[12], *b_v3 =(const float*)d_in[13];
  const float* w_avatt=(const float*)d_in[14],*b_avatt=(const float*)d_in[15];
  const float* w_a1=(const float*)d_in[16], *b_a1=(const float*)d_in[17];
  const float* w_v1=(const float*)d_in[18], *b_v1=(const float*)d_in[19];
  const float* w_bn=(const float*)d_in[20], *b_bn=(const float*)d_in[21];
  const float* w_ca=(const float*)d_in[22], *b_ca=(const float*)d_in[23];
  const float* w_v2=(const float*)d_in[24], *b_v2=(const float*)d_in[25];
  const float* w_a2=(const float*)d_in[26], *b_a2=(const float*)d_in[27];
  const float* w_sa=(const float*)d_in[28], *b_sa=(const float*)d_in[29];

  float* outCS    = (float*)d_out;                 // [BT, VD]
  float* audioOut = outCS + (long)BT_ * VD_;       // [BT, AD] == output #2 & GEMM input

  char* ws = (char*)d_ws;
  const size_t SZ_Q  = (size_t)TOK * QD_ * 4;      // 64225280
  const size_t SZ_V  = (size_t)TOK * VD_ * 4;      // 256901120
  const size_t SZ_V2 = (size_t)TOK * VD_ * 2;      // 128450560 (bf16)
  const size_t SZ_BV = (size_t)BT_ * VD_ * 4;      // 5242880
  const size_t SZ_BH = (size_t)BT_ * HD_ * 4;      // 2621440
  size_t o = 0;
  float*  Qb     = (float*)(ws + o);  o += SZ_Q;
  float*  Kbuf   = (float*)(ws + o);  o += SZ_Q;
  float*  Vb     = (float*)(ws + o);  o += SZ_V;
  __bf16* vf2    = (__bf16*)(ws + o); o += SZ_V2;
  float*  vf2avg = (float*)(ws + o);  o += SZ_BV;
  float*  vavg   = (float*)(ws + o);  o += SZ_BH;
  float*  aq1    = (float*)(ws + o);  o += SZ_BV;
  float*  meanb  = (float*)(ws + o);  o += SZ_BV;
  float*  avqb   = (float*)(ws + o);  o += SZ_BH;
  float*  chb    = (float*)(ws + o);  o += SZ_BV;
  float*  aq2    = (float*)(ws + o);  o += SZ_BH;
  float*  s_self = (float*)(ws + o);  o += (size_t)TOK * 4;
  float*  s_sp   = (float*)(ws + o);  o += (size_t)TOK * 4;
  // region reuse (dead after attn_ln): relu3->Q, cq->K, prod/catt->V halves
  __bf16* relu3  = (__bf16*)Qb;
  __bf16* cqb    = (__bf16*)Kbuf;
  __bf16* prod   = (__bf16*)Vb;
  __bf16* cattb  = (__bf16*)((char*)Vb + SZ_V2);

  dim3 blk(256);
  // 1) audio transpose (also second output)
  audio_tr_kernel<<<(BT_ * AD_) / 256, blk, 0, stream>>>(audio, audioOut);
  // 2-4) Q, K, V GEMMs (f32 out)
  gemm_wmma<false,false,0,false><<<dim3(QD_/64, TOK/128), blk, 0, stream>>>(video, wq, bq, Qb, nullptr, TOK, QD_, VD_);
  gemm_wmma<false,false,0,false><<<dim3(QD_/64, TOK/128), blk, 0, stream>>>(video, wk, bk, Kbuf, nullptr, TOK, QD_, VD_);
  gemm_wmma<false,false,0,false><<<dim3(VD_/64, TOK/128), blk, 0, stream>>>(video, wv, bv, Vb, nullptr, TOK, VD_, VD_);
  // 5) attention + LN
  size_t shmem = (size_t)(NSP*QD_*2 + NSP*VD_ + NSP*50 + 256) * 4;
  attn_ln_kernel<<<BT_, blk, shmem, stream>>>(video, Qb, Kbuf, Vb, ln_g, ln_b, vf2, vf2avg);
  // 6) video_avg = relu((vf2 mean) @ w_ave^T)
  gemm_wmma<false,false,1,false><<<dim3(HD_/64, BT_/128), blk, 0, stream>>>(vf2avg, w_ave, b_ave, vavg, nullptr, BT_, HD_, VD_);
  // 7) aq1 = relu(audio @ w_a1^T)
  gemm_wmma<false,false,1,false><<<dim3(VD_/64, BT_/128), blk, 0, stream>>>(audioOut, w_a1, b_a1, aq1, nullptr, BT_, VD_, AD_);
  // 8) relu3 = relu(vf2 @ w_v3^T) (bf16)
  gemm_wmma<true,true,1,false><<<dim3(HD_/64, TOK/128), blk, 0, stream>>>(vf2, w_v3, b_v3, relu3, nullptr, TOK, HD_, VD_);
  // 9) prod = relu(vf2 @ w_v1^T) * aq1[bt]  (bf16)
  gemm_wmma<true,true,1,true><<<dim3(VD_/64, TOK/128), blk, 0, stream>>>(vf2, w_v1, b_v1, prod, aq1, TOK, VD_, VD_);
  // 10) mean over 49
  mean49_kernel<<<(BT_ * VD_) / 256, blk, 0, stream>>>(prod, meanb);
  // 11) avq = relu(meanb @ w_bn^T)
  gemm_wmma<false,false,1,false><<<dim3(HD_/64, BT_/128), blk, 0, stream>>>(meanb, w_bn, b_bn, avqb, nullptr, BT_, HD_, VD_);
  // 12) ch = sigmoid(avq @ w_ca^T)
  gemm_wmma<false,false,2,false><<<dim3(VD_/64, BT_/128), blk, 0, stream>>>(avqb, w_ca, b_ca, chb, nullptr, BT_, VD_, HD_);
  // 13) c_att bf16
  catt_kernel<<<((long)TOK * VD_) / 256, blk, 0, stream>>>(video, chb, cattb);
  // 14) aq2 = relu(audio @ w_a2^T)
  gemm_wmma<false,false,1,false><<<dim3(HD_/64, BT_/128), blk, 0, stream>>>(audioOut, w_a2, b_a2, aq2, nullptr, BT_, HD_, AD_);
  // 15) cq = relu(c_att @ w_v2^T) (bf16)
  gemm_wmma<true,true,1,false><<<dim3(HD_/64, TOK/128), blk, 0, stream>>>(cattb, w_v2, b_v2, cqb, nullptr, TOK, HD_, VD_);
  // 16) scores
  score_kernel<<<TOK / 256, blk, 0, stream>>>(relu3, vavg, w_avatt, b_avatt,
                                              cqb, aq2, w_sa, b_sa, s_self, s_sp);
  // 17) final
  final_kernel<<<BT_, blk, 0, stream>>>(video, chb, vf2, s_self, s_sp, outCS);
}
